// SwarmAttentionWrapper_38027640439256
// MI455X (gfx1250) — matrix operand order repaired and checked by reading, not compiled, not run on hardware
//
#include <hip/hip_runtime.h>
#include <hip/hip_bf16.h>

// ---------------------------------------------------------------------------
// Problem constants (B=2, L=2048, H=16, d=64, D=1024, window=128)
// ---------------------------------------------------------------------------
#define MTOK   4096      // B*L
#define DMODEL 1024
#define N3     3072
#define SEQL   2048
#define NHEAD  16
#define HDIM   64
#define WIN    128

typedef __bf16 bf16x16 __attribute__((ext_vector_type(16)));
typedef float  f32x8   __attribute__((ext_vector_type(8)));

union FragB16 {
    unsigned int   u[8];
    unsigned short h[16];
    bf16x16        v;
};

__device__ __forceinline__ unsigned short f2bf(float f) {
    union { float f; unsigned u; } x; x.f = f;
    unsigned r = x.u + 0x7fffu + ((x.u >> 16) & 1u);   // round-to-nearest-even
    return (unsigned short)(r >> 16);
}

__device__ __forceinline__ unsigned short load_bf16(const float* p)          { return f2bf(*p); }
__device__ __forceinline__ unsigned short load_bf16(const unsigned short* p) { return *p; }
__device__ __forceinline__ void store_c(float* p, float v)                   { *p = v; }
__device__ __forceinline__ void store_c(unsigned short* p, float v)          { *p = f2bf(v); }

// LDS byte offset of a __shared__ object (generic -> AS(3) cast; per VGLOBAL
// async encoding, VDST holds the per-lane LDS byte address).
typedef __attribute__((address_space(3))) unsigned char lds_byte_t;
__device__ __forceinline__ unsigned lds_addr(void* p) {
    return (unsigned)(unsigned long long)(lds_byte_t*)p;
}

// ---------------------------------------------------------------------------
// Zero-fill (attention-weights region of d_out; masked entries must be 0)
// ---------------------------------------------------------------------------
__global__ void zero_f4(float4* __restrict__ p, long long n4) {
    long long i      = (long long)blockIdx.x * blockDim.x + threadIdx.x;
    long long stride = (long long)gridDim.x * blockDim.x;
    float4 z = make_float4(0.f, 0.f, 0.f, 0.f);
    for (; i < n4; i += stride) p[i] = z;
}

// ---------------------------------------------------------------------------
// GEMM: C[M,N] = A[M,K] @ B[K,N] + bias  (bf16 WMMA, f32 accumulate)
// BM=BN=64, BK=32; 256 threads = 8 waves, each wave owns two 16x16 tiles.
// ---------------------------------------------------------------------------
template <typename AT, typename CT>
__global__ __launch_bounds__(256)
void gemm_bias_wmma(const AT* __restrict__ A, const float* __restrict__ B,
                    const float* __restrict__ bias, CT* __restrict__ C,
                    int M, int N, int K) {
    __shared__ unsigned short As[64][32];   // [m][k] bf16
    __shared__ unsigned short Bs[64][32];   // [n][k] bf16 (B transposed)

    const int t    = threadIdx.x;
    const int lane = t & 31;
    const int wave = t >> 5;
    const int hi   = (lane >= 16) ? 1 : 0;
    const int ln   = lane & 15;

    const int bm = blockIdx.y * 64;
    const int bn = blockIdx.x * 64;
    const int tm = (wave >> 1) * 16;        // row tile within block
    const int tn = (wave & 1) * 32;         // col tile pair within block

    f32x8 acc0 = {}, acc1 = {};

    const int am  = t >> 2, ak  = (t & 3) * 8;   // A-tile loader mapping
    const int bk  = t >> 3, bnq = (t & 7) * 8;   // B-tile loader mapping

    for (int k0 = 0; k0 < K; k0 += 32) {
        // ---- stage A tile (convert to bf16) ----
        {
            const AT* src = A + (size_t)(bm + am) * K + (k0 + ak);
            union { uint4 q; unsigned short s[8]; } ta;
            #pragma unroll
            for (int j = 0; j < 8; ++j) ta.s[j] = load_bf16(src + j);
            *(uint4*)&As[am][ak] = ta.q;
            if (k0 + 32 < K) __builtin_prefetch(src + 32, 0, 0);   // global_prefetch_b8
        }
        // ---- stage B tile transposed (convert to bf16) ----
        {
            const float* src = B + (size_t)(k0 + bk) * N + (bn + bnq);
            #pragma unroll
            for (int j = 0; j < 8; ++j) Bs[bnq + j][bk] = f2bf(src[j]);
            if (k0 + 32 < K) __builtin_prefetch(src + (size_t)32 * N, 0, 0);
        }
        __syncthreads();

        // ---- fragment gather per ISA wave32 layouts ----
        FragB16 a, b0, b1;
        #pragma unroll
        for (int p = 0; p < 8; ++p) {   // A 16x32: K split {0..7, 16..23} (+8 for hi half)
            int kk = ((p < 4) ? 2 * p : 2 * p + 8) + hi * 8;
            a.u[p] = *(const unsigned int*)&As[tm + ln][kk];
        }
        #pragma unroll
        for (int p = 0; p < 8; ++p) {   // B 32x16: K split {0..7, 16..23} (+8 for hi half)
            int kk = ((p < 4) ? 2 * p : 2 * p + 8) + hi * 8;
            b0.u[p] = *(const unsigned int*)&Bs[tn + ln][kk];
            b1.u[p] = *(const unsigned int*)&Bs[tn + 16 + ln][kk];
        }
        acc0 = __builtin_amdgcn_wmma_f32_16x16x32_bf16(false, a.v, false, b0.v,
                                                       (short)0, acc0, false, false);
        acc1 = __builtin_amdgcn_wmma_f32_16x16x32_bf16(false, a.v, false, b1.v,
                                                       (short)0, acc1, false, false);
        __syncthreads();
    }

    // ---- epilogue: C/D layout -> lane ln = col, VGPR r = row (+8 for hi half) ----
    const int row0 = bm + tm + hi * 8;
    const int c0   = bn + tn + ln;
    const int c1   = c0 + 16;
    const float bs0 = bias[c0], bs1 = bias[c1];
    #pragma unroll
    for (int r = 0; r < 8; ++r) {
        store_c(C + (size_t)(row0 + r) * N + c0, acc0[r] + bs0);
        store_c(C + (size_t)(row0 + r) * N + c1, acc1[r] + bs1);
    }
}

// ---------------------------------------------------------------------------
// Sliding-window attention. One wave per (b, h, 16-query tile).
// Scores: 9 key tiles x (2 WMMA over d=64). Softmax in LDS. P·V: 5x4 WMMA.
// Key window covered: [q0-128, q0+16) padded to K=160 with zero probabilities.
// V tile streams global->LDS via GLOBAL_LOAD_ASYNC_TO_LDS_B128 (ASYNCcnt),
// fully overlapped with the Q·K^T WMMAs; resolved by s_wait_asynccnt.
// ---------------------------------------------------------------------------
__global__ __launch_bounds__(32)
void sliding_attn_wmma(const unsigned short* __restrict__ qkv,   // [4096][3072] bf16
                       float* __restrict__ attnw,                // [2][16][2048][2048] f32
                       unsigned short* __restrict__ ctx)         // [4096][1024] bf16
{
    __shared__ float          S[16][160];     // raw scores
    __shared__ unsigned short P[16][160];     // probabilities, bf16
    __shared__ unsigned short Vs[160][64];    // V rows for the window

    const int lane = threadIdx.x & 31;
    const int hi   = (lane >= 16) ? 1 : 0;
    const int m    = lane & 15;
    const int qt   = blockIdx.x & 127;
    const int h    = (blockIdx.x >> 7) & 15;
    const int b    = blockIdx.x >> 11;
    const int q0   = qt * 16;

    // ---- kick off async V-window staging: 160 rows x 128B = 1280 16B chunks,
    //      one chunk per lane per issue, 40 issues, no VGPR data path ----
    {
        const unsigned vs_base = lds_addr(&Vs[0][0]);
        #pragma unroll
        for (int it = 0; it < 40; ++it) {
            int c   = it * 32 + lane;            // chunk index
            int row = c >> 3;                    // 0..159 within window
            int kv  = q0 - WIN + row;
            int kvc = kv < 0 ? 0 : (kv > SEQL - 1 ? SEQL - 1 : kv);
            unsigned long long ga = (unsigned long long)(const void*)
                (qkv + ((size_t)(b * SEQL + kvc)) * N3 + 2 * DMODEL + h * HDIM + (c & 7) * 8);
            unsigned ldst = vs_base + (unsigned)c * 16u;
            asm volatile("global_load_async_to_lds_b128 %0, %1, off"
                         :: "v"(ldst), "v"(ga) : "memory");
        }
    }

    // ---- Q fragments (A layout 16x32, two frags cover d=64), vectorized b128 ----
    FragB16 qf0, qf1;
    {
        const uint4* q4 = (const uint4*)(qkv + ((size_t)(b * SEQL + q0 + m)) * N3 + h * HDIM);
        ((uint4*)qf0.u)[0] = q4[hi];        // d =  8*hi .. +7   (e0..7)
        ((uint4*)qf0.u)[1] = q4[hi + 2];    // d = 16+8*hi .. +7 (e8..15)
        ((uint4*)qf1.u)[0] = q4[hi + 4];
        ((uint4*)qf1.u)[1] = q4[hi + 6];
    }

    // ---- pad score columns 144..159 ----
    for (int i = lane; i < 256; i += 32) S[i >> 4][144 + (i & 15)] = -1e30f;

    // ---- scores: S = (Q @ K^T) / sqrt(d), masked sliding-window causal ----
    #pragma unroll
    for (int j = 0; j < 9; ++j) {
        int ks = q0 + (j - 8) * 16;                 // uniform across wave
        if (ks >= 0) {
            FragB16 kf0, kf1;
            const uint4* k4 = (const uint4*)(qkv + ((size_t)(b * SEQL + ks + m)) * N3 + DMODEL + h * HDIM);
            ((uint4*)kf0.u)[0] = k4[hi];            // d = 8*hi .. +7, 16+8*hi .. +7
            ((uint4*)kf0.u)[1] = k4[hi + 2];
            ((uint4*)kf1.u)[0] = k4[hi + 4];
            ((uint4*)kf1.u)[1] = k4[hi + 6];
            f32x8 s = {};
            s = __builtin_amdgcn_wmma_f32_16x16x32_bf16(false, qf0.v, false, kf0.v,
                                                        (short)0, s, false, false);
            s = __builtin_amdgcn_wmma_f32_16x16x32_bf16(false, qf1.v, false, kf1.v,
                                                        (short)0, s, false, false);
            int gk = ks + m;
            #pragma unroll
            for (int r = 0; r < 8; ++r) {
                int gq   = q0 + r + 8 * hi;
                int dist = gq - gk;
                float v  = (dist >= 0 && dist < WIN) ? s[r] * 0.125f : -1e30f;
                S[r + 8 * hi][j * 16 + m] = v;
            }
        } else {
            #pragma unroll
            for (int r = 0; r < 8; ++r) S[r + 8 * hi][j * 16 + m] = -1e30f;
        }
    }

    // ---- V window must be resident in LDS before the PV stage ----
    asm volatile("s_wait_asynccnt 0x0" ::: "memory");
    __syncthreads();

    // ---- softmax: 2 lanes per row, pair-combine via shfl_xor ----
    {
        const int row = lane >> 1;
        const int c0  = (lane & 1) * 80;
        float mx = -1e30f;
        for (int c = 0; c < 80; ++c) mx = fmaxf(mx, S[row][c0 + c]);
        mx = fmaxf(mx, __shfl_xor(mx, 1, 32));
        float sum = 0.f;
        for (int c = 0; c < 80; ++c) sum += __expf(S[row][c0 + c] - mx);
        sum += __shfl_xor(sum, 1, 32);
        float inv = 1.f / sum;
        float* wrow = attnw + (((size_t)(b * NHEAD + h) * SEQL) + (q0 + row)) * SEQL;
        for (int c = 0; c < 80; ++c) {
            int cc  = c0 + c;
            float p = __expf(S[row][cc] - mx) * inv;
            P[row][cc] = f2bf(p);
            int key = q0 - WIN + cc;                 // cc<144 => key <= q0+15 < SEQL
            if (cc < 144 && key >= 0) wrow[key] = p;
        }
    }
    __syncthreads();

    // ---- O = P(16x160) @ V(160x64): 5 K-steps x 4 d-tiles of WMMA ----
    f32x8 o[4] = {{}, {}, {}, {}};
    #pragma unroll
    for (int kst = 0; kst < 5; ++kst) {
        FragB16 pa;
        #pragma unroll
        for (int p = 0; p < 8; ++p) {
            int kk = kst * 32 + ((p < 4) ? 2 * p : 2 * p + 8) + 8 * hi;
            pa.u[p] = *(const unsigned int*)&P[m][kk];
        }
        #pragma unroll
        for (int dt = 0; dt < 4; ++dt) {
            FragB16 vb;
            #pragma unroll
            for (int e = 0; e < 16; ++e) {
                int key = kst * 32 + ((e < 8) ? e : e + 8) + 8 * hi;
                vb.h[e] = Vs[key][dt * 16 + m];
            }
            o[dt] = __builtin_amdgcn_wmma_f32_16x16x32_bf16(false, pa.v, false, vb.v,
                                                            (short)0, o[dt], false, false);
        }
    }

    // ---- write context (bf16) back in [tok][H*d] layout for the proj GEMM ----
    #pragma unroll
    for (int dt = 0; dt < 4; ++dt) {
        #pragma unroll
        for (int r = 0; r < 8; ++r) {
            int tok = b * SEQL + q0 + r + 8 * hi;
            ctx[(size_t)tok * DMODEL + h * HDIM + dt * 16 + m] = f2bf(o[dt][r]);
        }
    }
}

// ---------------------------------------------------------------------------
// Launch
// ---------------------------------------------------------------------------
extern "C" void kernel_launch(void* const* d_in, const int* in_sizes, int n_in,
                              void* d_out, int out_size, void* d_ws, size_t ws_size,
                              hipStream_t stream) {
    (void)in_sizes; (void)n_in; (void)out_size; (void)ws_size;

    const float* hidden = (const float*)d_in[0];
    const float* Wqkv   = (const float*)d_in[1];
    const float* bqkv   = (const float*)d_in[2];
    const float* Wproj  = (const float*)d_in[3];
    const float* bproj  = (const float*)d_in[4];
    // d_in[5] = window_size (128; geometry specialized for it)

    float* out   = (float*)d_out;
    float* attnw = out + (size_t)MTOK * DMODEL;          // [2,16,2048,2048]

    unsigned short* qkv = (unsigned short*)d_ws;                   // bf16 [4096][3072]
    unsigned short* ctx = qkv + (size_t)MTOK * N3;                 // bf16 [4096][1024]

    // 1) zero the attention-weight output (masked entries stay 0)
    long long n4 = (long long)2 * NHEAD * SEQL * SEQL / 4;
    zero_f4<<<8192, 256, 0, stream>>>((float4*)attnw, n4);

    // 2) QKV projection (fp32 in -> bf16 out)
    gemm_bias_wmma<float, unsigned short>
        <<<dim3(N3 / 64, MTOK / 64), 256, 0, stream>>>(hidden, Wqkv, bqkv, qkv,
                                                       MTOK, N3, DMODEL);

    // 3) sliding-window attention (writes attnw windows + bf16 context)
    sliding_attn_wmma<<<2 * NHEAD * (SEQL / 16), 32, 0, stream>>>(qkv, attnw, ctx);

    // 4) output projection (bf16 in -> fp32 out)
    gemm_bias_wmma<unsigned short, float>
        <<<dim3(DMODEL / 64, MTOK / 64), 256, 0, stream>>>(ctx, Wproj, bproj, out,
                                                           MTOK, DMODEL, DMODEL);
}
